// MultiHeadAttLoc_17884243821373
// MI455X (gfx1250) — matrix operand order repaired and checked by reading, not compiled, not run on hardware
//
#include <hip/hip_runtime.h>
#include <math.h>

// Problem dims (fixed by the reference)
#define Hh 4
#define Bb 16
#define Tt 4000
#define Ee 512
#define Dd 512
#define Aa 128
#define Cc 10
#define Kc 101
#define Oo 512

typedef __attribute__((ext_vector_type(16))) __bf16 v16bf;
typedef __attribute__((ext_vector_type(8)))  float  v8f;

union Frag16 { v16bf v; unsigned int u[8]; };
union FragAcc { v8f v; float f[8]; };

__device__ __forceinline__ unsigned int pack_bf16(float a, float b) {
  unsigned int ua = __float_as_uint(a);
  unsigned int ub = __float_as_uint(b);
  ua = (ua + 0x7FFFu + ((ua >> 16) & 1u)) >> 16;   // round-to-nearest-even
  ub = (ub + 0x7FFFu + ((ub >> 16) & 1u)) >> 16;
  return (ua & 0xFFFFu) | ((ub & 0xFFFFu) << 16);
}

// ---------------------------------------------------------------------------
// Pack [Wenc | Watt] into B-fragment order (bf16), 17 k-tiles of 32 per
// (head, n-tile). Fragment = 32x16 bf16 = 256 dwords, lane L owns dwords
// [L*8, L*8+8): col n = L%16, rows k = (L/16)*16 + 2j,2j+1.
// ---------------------------------------------------------------------------
__global__ void prep_fragW(const float* __restrict__ Wenc,
                           const float* __restrict__ Watt,
                           unsigned int* __restrict__ fragW) {
  int d = blockIdx.x * blockDim.x + threadIdx.x;
  const int NDW = Hh * 8 * 17 * 256;
  if (d >= NDW) return;
  int j  = d & 7;
  int L  = (d >> 3) & 31;
  int f  = d >> 8;
  int kt = f % 17;
  int r  = f / 17;
  int nt8 = r & 7;
  int h   = r >> 3;
  int n  = (nt8 << 4) + (L & 15);        // a index within head
  int k0 = ((L >> 4) << 4) + (j << 1);   // k row pair base (0..30)
  float x0, x1;
  if (kt < 16) {
    int e0 = kt * 32 + k0;
    x0 = Wenc[(h * Ee + e0) * Aa + n];
    x1 = Wenc[(h * Ee + e0 + 1) * Aa + n];
  } else {  // Watt k-extension, rows = conv channels (10 valid, rest zero)
    x0 = (k0     < Cc) ? Watt[(h * Cc + k0    ) * Aa + n] : 0.f;
    x1 = (k0 + 1 < Cc) ? Watt[(h * Cc + k0 + 1) * Aa + n] : 0.f;
  }
  fragW[d] = pack_bf16(x0, x1);
}

// dproj[b,h,a] = benc[h,a] + sum_d dec_z[b,d]*Wdec[h,d,a]
__global__ void prep_dproj(const float* __restrict__ dec_z,
                           const float* __restrict__ Wdec,
                           const float* __restrict__ benc,
                           float* __restrict__ dproj) {
  int g = blockIdx.x * blockDim.x + threadIdx.x;
  if (g >= Bb * Hh * Aa) return;
  int a = g & (Aa - 1);
  int h = (g >> 7) & (Hh - 1);
  int b = g >> 9;
  float s = benc[h * Aa + a];
  for (int dd = 0; dd < Dd; ++dd)
    s += dec_z[b * Dd + dd] * Wdec[(h * Dd + dd) * Aa + a];
  dproj[g] = s;
}

// Location conv: convP[b,h,t, c(16, 10 valid)] in bf16 pairs (8 dwords/row)
__global__ void prep_conv(const float* __restrict__ att_prev,
                          const float* __restrict__ conv_w,
                          unsigned int* __restrict__ convP_u) {
  int g = blockIdx.x * blockDim.x + threadIdx.x;  // over B*H*T
  if (g >= Bb * Hh * Tt) return;
  int t = g % Tt;
  int h = (g / Tt) % Hh;
  int b = g / (Tt * Hh);
  float acc[Cc];
#pragma unroll
  for (int c = 0; c < Cc; ++c) acc[c] = 0.f;
  const float* xrow = att_prev + (size_t)(h * Bb + b) * Tt;
  for (int k = 0; k < Kc; ++k) {
    int ts = t + k - (Kc / 2);
    if (ts >= 0 && ts < Tt) {
      float x = xrow[ts];
#pragma unroll
      for (int c = 0; c < Cc; ++c)
        acc[c] += conv_w[(h * Cc + c) * Kc + k] * x;
    }
  }
  unsigned int* outp = convP_u + (size_t)g * 8;
#pragma unroll
  for (int p = 0; p < 8; ++p) {
    float a0 = (2 * p     < Cc) ? acc[2 * p]     : 0.f;
    float a1 = (2 * p + 1 < Cc) ? acc[2 * p + 1] : 0.f;
    outp[p] = pack_bf16(a0, a1);
  }
}

// ---------------------------------------------------------------------------
// Main fused kernel: e[b,h,t] = sum_a tanh(enc@Wenc + conv@Watt + dproj)*gv
// bf16 WMMA 16x16x32, K = 512(enc) + 32(conv pad) = 17 k-tiles.
// Wave handles 16 t-rows, loops 32 n-tiles (4 heads x 8 tiles of A=128).
// A fragments live in registers for the whole n loop (enc read once).
// ---------------------------------------------------------------------------
__global__ __launch_bounds__(256)
void attn_energy_wmma(const float* __restrict__ enc,
                      const unsigned int* __restrict__ fragW,
                      const unsigned int* __restrict__ convP,
                      const float* __restrict__ dproj,
                      const float* __restrict__ gv,
                      float* __restrict__ e_buf) {
  const int b    = blockIdx.y;
  const int lane = threadIdx.x & 31;
  const int wv   = threadIdx.x >> 5;
  const int t0   = blockIdx.x * 128 + wv * 16;
  const int m    = lane & 15;
  const int trow = t0 + m;
  const int tc   = trow < Tt ? trow : (Tt - 1);
  const int kh   = (lane < 16) ? 0 : 16;  // A-frag K half select

  // A fragments: 16 k-tiles of the enc row slice, f32 -> bf16 on the fly.
  Frag16 afr[16];
  const float* rowp = enc + ((size_t)b * Tt + tc) * Ee;
#pragma unroll
  for (int kt = 0; kt < 16; ++kt) {
#pragma unroll
    for (int v = 0; v < 8; ++v) {
      int k = (v << 1) + kh;
      const float2 p = *(const float2*)(rowp + kt * 32 + k);
      afr[kt].u[v] = pack_bf16(p.x, p.y);
    }
  }

  float psum[8];
  Frag16 cf;

  for (int nt = 0; nt < 32; ++nt) {
    const int h   = nt >> 3;
    const int ntl = nt & 7;
    if (ntl == 0) {  // conv A-fragment for this head; reset row partials
      const unsigned int* cp = convP + ((size_t)(b * Hh + h) * Tt + tc) * 8;
#pragma unroll
      for (int v = 0; v < 8; ++v) {
        int k = (v << 1) + kh;
        cf.u[v] = (k < 16) ? cp[k >> 1] : 0u;
      }
#pragma unroll
      for (int r = 0; r < 8; ++r) psum[r] = 0.f;
    }

    FragAcc acc;
#pragma unroll
    for (int r = 0; r < 8; ++r) acc.f[r] = 0.f;

    const unsigned int* fb =
        fragW + ((((h << 3) + ntl) * 17) << 8) + (lane << 3);
#pragma unroll
    for (int kt = 0; kt < 17; ++kt) {
      Frag16 bf;
      const uint4* bp = (const uint4*)(fb + (kt << 8));
      uint4 b0 = bp[0];
      uint4 b1 = bp[1];
      bf.u[0] = b0.x; bf.u[1] = b0.y; bf.u[2] = b0.z; bf.u[3] = b0.w;
      bf.u[4] = b1.x; bf.u[5] = b1.y; bf.u[6] = b1.z; bf.u[7] = b1.w;
      const v16bf av = (kt < 16) ? afr[kt].v : cf.v;
      acc.v = __builtin_amdgcn_wmma_f32_16x16x32_bf16(
          false, av, false, bf.v, (short)0, acc.v, false, false);
    }

    // epilogue: + dproj(+benc), tanh, * gv; lane holds col a, rows r(+8)
    const int a  = (ntl << 4) + (lane & 15);
    const float dp = dproj[(b * Hh + h) * Aa + a];
    const float g  = gv[h * Aa + a];
#pragma unroll
    for (int r = 0; r < 8; ++r) {
      float y = tanhf(acc.f[r] + dp);
      psum[r] += y * g;
    }

    if (ntl == 7) {  // head done: reduce over the 16 cols held per half-wave
#pragma unroll
      for (int r = 0; r < 8; ++r) {
        float s = psum[r];
        s += __shfl_xor(s, 1, 32);
        s += __shfl_xor(s, 2, 32);
        s += __shfl_xor(s, 4, 32);
        s += __shfl_xor(s, 8, 32);
        psum[r] = s;
      }
      const int rbase = (lane < 16) ? 0 : 8;
      if (lane == 0 || lane == 16) {
#pragma unroll
        for (int r = 0; r < 8; ++r) {
          int t = t0 + rbase + r;
          if (t < Tt)
            e_buf[(size_t)(b * Hh + h) * Tt + t] = psum[r];
        }
      }
    }
  }
}

// Softmax over T for each (b,h); writes w in output layout [H,B,T].
__global__ __launch_bounds__(256)
void softmax_rows(const float* __restrict__ e_buf, float* __restrict__ wout) {
  __shared__ float red[256];
  __shared__ float stat[2];
  const int bh = blockIdx.x;  // = h*B + b
  const int h = bh >> 4;
  const int b = bh & 15;
  const float* er = e_buf + (size_t)(b * Hh + h) * Tt;
  const int tid = threadIdx.x;

  float m = -3.4e38f;
  for (int t = tid; t < Tt; t += 256) m = fmaxf(m, er[t]);
  red[tid] = m;
  __syncthreads();
  for (int s = 128; s > 0; s >>= 1) {
    if (tid < s) red[tid] = fmaxf(red[tid], red[tid + s]);
    __syncthreads();
  }
  if (tid == 0) stat[0] = red[0];
  __syncthreads();
  const float M = stat[0];

  float sum = 0.f;
  for (int t = tid; t < Tt; t += 256) sum += __expf(er[t] - M);
  __syncthreads();
  red[tid] = sum;
  __syncthreads();
  for (int s = 128; s > 0; s >>= 1) {
    if (tid < s) red[tid] += red[tid + s];
    __syncthreads();
  }
  if (tid == 0) stat[1] = red[0];
  __syncthreads();
  const float inv = 1.f / stat[1];

  float* wr = wout + (size_t)bh * Tt;
  for (int t = tid; t < Tt; t += 256) wr[t] = __expf(er[t] - M) * inv;
}

// Context partials: split T into 8 deterministic chunks (no float atomics).
__global__ __launch_bounds__(128)
void context_partial(const float* __restrict__ enc,
                     const float* __restrict__ wout,
                     float* __restrict__ cpart) {
  const int e = blockIdx.x * 128 + threadIdx.x;
  const int b = blockIdx.y;
  const int p = blockIdx.z;
  float a0 = 0.f, a1 = 0.f, a2 = 0.f, a3 = 0.f;
  const int tbeg = p * (Tt / 8), tend = tbeg + (Tt / 8);
  for (int t = tbeg; t < tend; ++t) {
    float x = enc[((size_t)b * Tt + t) * Ee + e];
    a0 += wout[(0 * Bb + b) * Tt + t] * x;
    a1 += wout[(1 * Bb + b) * Tt + t] * x;
    a2 += wout[(2 * Bb + b) * Tt + t] * x;
    a3 += wout[(3 * Bb + b) * Tt + t] * x;
  }
  float* cp = cpart + ((size_t)p * Bb + b) * (Hh * Ee);
  cp[0 * Ee + e] = a0;
  cp[1 * Ee + e] = a1;
  cp[2 * Ee + e] = a2;
  cp[3 * Ee + e] = a3;
}

__global__ void reduce_c(const float* __restrict__ cpart, float* __restrict__ c) {
  int g = blockIdx.x * blockDim.x + threadIdx.x;  // B*H*E
  if (g >= Bb * Hh * Ee) return;
  float s = 0.f;
#pragma unroll
  for (int p = 0; p < 8; ++p) s += cpart[(size_t)p * (Bb * Hh * Ee) + g];
  c[g] = s;
}

__global__ __launch_bounds__(256)
void out_gemm(const float* __restrict__ c, const float* __restrict__ Wo,
              const float* __restrict__ bo, float* __restrict__ out) {
  const int o = blockIdx.x * 256 + threadIdx.x;
  const int b = blockIdx.y;
  float s = bo[o];
  const float* cr = c + b * (Hh * Ee);
  for (int k = 0; k < Hh * Ee; ++k) s += cr[k] * Wo[(size_t)k * Oo + o];
  out[b * Oo + o] = s;
}

extern "C" void kernel_launch(void* const* d_in, const int* in_sizes, int n_in,
                              void* d_out, int out_size, void* d_ws, size_t ws_size,
                              hipStream_t stream) {
  (void)in_sizes; (void)n_in; (void)out_size; (void)ws_size;
  const float* enc      = (const float*)d_in[0];
  // d_in[1] = enc_len (unused by the reference)
  const float* dec_z    = (const float*)d_in[2];
  const float* att_prev = (const float*)d_in[3];
  const float* Wenc     = (const float*)d_in[4];
  const float* benc     = (const float*)d_in[5];
  const float* Wdec     = (const float*)d_in[6];
  const float* Watt     = (const float*)d_in[7];
  const float* conv_w   = (const float*)d_in[8];
  const float* gv       = (const float*)d_in[9];
  const float* Wo       = (const float*)d_in[10];
  const float* bo       = (const float*)d_in[11];

  float* out  = (float*)d_out;
  float* wout = out + Bb * Oo;  // second output: w as [H,B,T]

  char* ws = (char*)d_ws;
  unsigned int* fragW = (unsigned int*)(ws);             //   557,056 B
  unsigned int* convP = (unsigned int*)(ws + 557056);    // 8,192,000 B
  float* dproj        = (float*)(ws + 8749056);          //    32,768 B
  float* e_buf        = (float*)(ws + 8781824);          // 1,024,000 B
  float* cpart        = (float*)(ws + 9805824);          // 1,048,576 B
  float* cbuf         = (float*)(ws + 10854400);         //   131,072 B

  prep_fragW<<<(Hh * 8 * 17 * 256 + 255) / 256, 256, 0, stream>>>(Wenc, Watt, fragW);
  prep_dproj<<<(Bb * Hh * Aa + 255) / 256, 256, 0, stream>>>(dec_z, Wdec, benc, dproj);
  prep_conv<<<(Bb * Hh * Tt + 255) / 256, 256, 0, stream>>>(att_prev, conv_w, convP);

  attn_energy_wmma<<<dim3((Tt + 127) / 128, Bb), 256, 0, stream>>>(
      enc, fragW, convP, dproj, gv, e_buf);

  softmax_rows<<<Hh * Bb, 256, 0, stream>>>(e_buf, wout);
  context_partial<<<dim3(Ee / 128, Bb, 8), 128, 0, stream>>>(enc, wout, cpart);
  reduce_c<<<(Bb * Hh * Ee + 255) / 256, 256, 0, stream>>>(cpart, cbuf);
  out_gemm<<<dim3(Oo / 256, Bb), 256, 0, stream>>>(cbuf, Wo, bo, out);
}
